// MLPBlock_2619930051210
// MI455X (gfx1250) — matrix operand order repaired and checked by reading, not compiled, not run on hardware
//
#include <hip/hip_runtime.h>
#include <hip/hip_bf16.h>

typedef __bf16 bf16;
typedef bf16  v8bf  __attribute__((ext_vector_type(8)));
typedef bf16  v16bf __attribute__((ext_vector_type(16)));
typedef float v8f   __attribute__((ext_vector_type(8)));
typedef unsigned int u32x4 __attribute__((ext_vector_type(4)));
typedef int i32x4 __attribute__((ext_vector_type(4)));
typedef int i32x8 __attribute__((ext_vector_type(8)));

#define H_DIM 1024
#define I_DIM 2048
#define TWO_I 4096
#define E_NUM 16
#define B_NUM 1024
#define LIMIT 7.0f
#define EPS_V 1e-5f

// ---------------------------------------------------------------------------
// Kernel 1: RMSNorm (fp32) + gate logits + top-2 routing + expert list build
// ---------------------------------------------------------------------------
__global__ void __launch_bounds__(256) k_norm_gate(
    const bf16* __restrict__ x, const float* __restrict__ nscale,
    const bf16* __restrict__ gw, const bf16* __restrict__ gb,
    bf16* __restrict__ t, int* __restrict__ idx, float* __restrict__ wgt,
    int* __restrict__ ecount, int* __restrict__ elist)
{
    const int b = blockIdx.x, tid = threadIdx.x;
    const int lane = tid & 31, wave = tid >> 5;
    __shared__ float red[8];
    __shared__ float inv_s;
    __shared__ float gpart[8][E_NUM];
    __shared__ float logits[E_NUM];

    const int h0 = tid * 4;
    float xv[4];
#pragma unroll
    for (int j = 0; j < 4; ++j) xv[j] = (float)x[(size_t)b * H_DIM + h0 + j];
    float ss = xv[0]*xv[0] + xv[1]*xv[1] + xv[2]*xv[2] + xv[3]*xv[3];
    for (int o = 16; o; o >>= 1) ss += __shfl_down(ss, o, 32);
    if (lane == 0) red[wave] = ss;
    __syncthreads();
    if (tid == 0) {
        float tot = 0.f;
        for (int w = 0; w < 8; ++w) tot += red[w];
        inv_s = rsqrtf(tot / (float)H_DIM + EPS_V);
    }
    __syncthreads();
    const float inv = inv_s;

    float tv[4];
#pragma unroll
    for (int j = 0; j < 4; ++j) {
        tv[j] = xv[j] * inv * nscale[h0 + j];
        t[(size_t)b * H_DIM + h0 + j] = (bf16)tv[j];
    }

    float pe[E_NUM];
#pragma unroll
    for (int e = 0; e < E_NUM; ++e) {
        float s = 0.f;
#pragma unroll
        for (int j = 0; j < 4; ++j) s += tv[j] * (float)gw[(size_t)e * H_DIM + h0 + j];
        pe[e] = s;
    }
#pragma unroll
    for (int e = 0; e < E_NUM; ++e) {
        float s = pe[e];
        for (int o = 16; o; o >>= 1) s += __shfl_down(s, o, 32);
        if (lane == 0) gpart[wave][e] = s;
    }
    __syncthreads();
    if (tid < E_NUM) {
        float s = 0.f;
        for (int w = 0; w < 8; ++w) s += gpart[w][tid];
        logits[tid] = s + (float)gb[tid];
    }
    __syncthreads();
    if (tid == 0) {
        int i1 = 0; float v1 = logits[0];
        for (int e = 1; e < E_NUM; ++e) if (logits[e] > v1) { v1 = logits[e]; i1 = e; }
        int i2 = (i1 == 0) ? 1 : 0; float v2 = logits[i2];
        for (int e = 0; e < E_NUM; ++e)
            if (e != i1 && logits[e] > v2) { v2 = logits[e]; i2 = e; }
        float w1 = 1.0f / (1.0f + __expf(v2 - v1));   // softmax over {v1,v2}
        float w2 = 1.0f - w1;
        idx[b * 2] = i1; idx[b * 2 + 1] = i2;
        wgt[b * 2] = w1; wgt[b * 2 + 1] = w2;
        int p1 = atomicAdd(&ecount[i1], 1); elist[i1 * B_NUM + p1] = b * 2;
        int p2 = atomicAdd(&ecount[i2], 1); elist[i2 * B_NUM + p2] = b * 2 + 1;
    }
}

// ---------------------------------------------------------------------------
// Kernel 2: grouped GEMM (MLP1) + interleaved clamped SwiGLU.
// 32-token M-tile per block: rows 0-15 gathered into LDS by the TDM
// (tensor_load_to_lds, gather mode), rows 16-31 read per-lane from global.
// Each wave holds two accumulators sharing one weight (B) fragment:
// 2 WMMAs per 128B weight load -> half the mlp1_w stream.
// ---------------------------------------------------------------------------
__global__ void __launch_bounds__(256) k_mlp1_swiglu(
    const bf16* __restrict__ t, const bf16* __restrict__ w1,
    const bf16* __restrict__ b1, const float* __restrict__ wgt,
    const int* __restrict__ ecount, const int* __restrict__ elist,
    bf16* __restrict__ actw)
{
    const int e    = blockIdx.z;
    const int cnt  = ecount[e];
    const int base = blockIdx.y * 32;
    if (base >= cnt) return;

    const int tid = threadIdx.x, lane = tid & 31, wave = tid >> 5;
    __shared__ bf16 lt[16 * H_DIM];   // 32 KB: token rows 0..15 (TDM gather dest)
    __shared__ int  rows[32];
    if (tid < 32) {
        int p = base + tid; if (p > cnt - 1) p = cnt - 1;
        rows[tid] = elist[e * B_NUM + p];
    }

    if (wave == 0) {
        // --- Tensor Data Mover: gather 16 rows of t[token,:] into LDS ------
        unsigned ldsa = (unsigned)(unsigned long long)(uintptr_t)&lt[0];
        unsigned long long ga = (unsigned long long)(uintptr_t)t;
        u32x4 g0;
        g0[0] = (1u << 31) | 1u;                 // gather_mode=1, 16b idx, count=1
        g0[1] = ldsa;                            // lds_addr
        g0[2] = (unsigned)(ga & 0xffffffffu);    // global_addr[31:0]
        g0[3] = ((unsigned)(ga >> 32) & 0x01ffffffu) | (2u << 30); // addr[56:32], type=2
        i32x8 g1;
        g1[0] = 1 << 16;                         // data_size = 2 bytes
        g1[1] = (H_DIM & 0xffff) << 16;          // tensor_dim0 = H (lo16)
        g1[2] = (B_NUM & 0xffff) << 16;          // tensor_dim1 = B (lo16)
        g1[3] = (H_DIM & 0xffff) << 16;          // tile_dim0 = H
        g1[4] = 16;                              // tile_dim1 = #valid indices
        g1[5] = H_DIM;                           // tensor_dim0_stride
        g1[6] = 0;
        g1[7] = 0;
        int toks[16];
#pragma unroll
        for (int j = 0; j < 16; ++j) {           // uniform -> SGPR values
            int p = base + j; if (p > cnt - 1) p = cnt - 1;
            toks[j] = elist[e * B_NUM + p] >> 1;
        }
        i32x4 g2, g3;
#pragma unroll
        for (int d = 0; d < 4; ++d) {
            g2[d] = (toks[2*d]     & 0xffff) | (toks[2*d + 1] << 16);
            g3[d] = (toks[8 + 2*d] & 0xffff) | (toks[9 + 2*d] << 16);
        }
        i32x8 gz = {};                           // extra descriptor group (unused)
        __builtin_amdgcn_tensor_load_to_lds(g0, g1, g2, g3, gz, 0);
        __builtin_amdgcn_s_wait_tensorcnt(0);
    }
    __syncthreads();

    const int n0  = blockIdx.x * 128 + wave * 16;  // column base in [0, 2I)
    const int col = lane & 15, hi = lane >> 4, m = lane & 15;
    const bf16* wrow  = w1 + (size_t)e * TWO_I * H_DIM + (size_t)(n0 + col) * H_DIM;
    const bf16* trow1 = t + (size_t)(rows[16 + m] >> 1) * H_DIM;  // tile-1 A rows

    v8f acc0 = {}, acc1 = {};
    for (int k0 = 0; k0 < H_DIM; k0 += 32) {
        union { v16bf v; v8bf h[2]; } A0, A1, Bm;
        // A (16x32, M=lane%16): lanes<16 hold K {0..7,16..23}, lanes>=16 {8..15,24..31}
        A0.h[0] = *(const v8bf*)&lt[m * H_DIM + k0 + hi * 8];
        A0.h[1] = *(const v8bf*)&lt[m * H_DIM + k0 + 16 + hi * 8];
        A1.h[0] = *(const v8bf*)&trow1[k0 + hi * 8];
        A1.h[1] = *(const v8bf*)&trow1[k0 + 16 + hi * 8];
        // B (32x16, N=lane%16): lanes<16 hold K {0..7,16..23}, lanes>=16 {8..15,24..31} of W1 row (n0+col)
        Bm.h[0] = *(const v8bf*)&wrow[k0 + hi * 8];
        Bm.h[1] = *(const v8bf*)&wrow[k0 + 16 + hi * 8];
        acc0 = __builtin_amdgcn_wmma_f32_16x16x32_bf16(
                   false, A0.v, false, Bm.v, (short)0, acc0, false, false);
        acc1 = __builtin_amdgcn_wmma_f32_16x16x32_bf16(
                   false, A1.v, false, Bm.v, (short)0, acc1, false, false);
    }

    const float bias = (float)b1[e * TWO_I + n0 + col];
    const bool even = (lane & 1) == 0;   // C col = lane%16 -> glu/lin in lane pairs
#pragma unroll
    for (int tile = 0; tile < 2; ++tile) {
        v8f acc = tile ? acc1 : acc0;
#pragma unroll
        for (int v = 0; v < 8; ++v) {
            float self  = acc[v] + bias;
            float other = __shfl_xor(self, 1, 32);
            if (even) {
                int mrow = tile * 16 + v + hi * 8;   // row within 32-token tile
                if (base + mrow < cnt) {
                    float glu = fminf(self, LIMIT);
                    float lin = fminf(fmaxf(other, -LIMIT), LIMIT);
                    float a   = glu / (1.0f + __expf(-1.702f * glu)) * (lin + 1.0f);
                    int entry = rows[mrow];          // == token*2 + slot
                    float wv  = wgt[entry];          // fold routing weight here
                    actw[(size_t)entry * I_DIM + ((n0 + col) >> 1)] = (bf16)(a * wv);
                }
            }
        }
    }
}

// ---------------------------------------------------------------------------
// Kernel 3: grouped GEMM (MLP2), 32-token M-tile, two accumulators per wave
// sharing the weight fragment; fp32 atomic scatter into accumulator.
// ---------------------------------------------------------------------------
__global__ void __launch_bounds__(256) k_mlp2(
    const bf16* __restrict__ actw, const bf16* __restrict__ w2,
    const int* __restrict__ ecount, const int* __restrict__ elist,
    float* __restrict__ out_acc)
{
    const int e    = blockIdx.z;
    const int cnt  = ecount[e];
    const int base = blockIdx.y * 32;
    if (base >= cnt) return;

    const int tid = threadIdx.x, lane = tid & 31, wave = tid >> 5;
    __shared__ int rows[32];
    if (tid < 32) {
        int p = base + tid; if (p > cnt - 1) p = cnt - 1;
        rows[tid] = elist[e * B_NUM + p];
    }
    __syncthreads();

    const int n0  = blockIdx.x * 128 + wave * 16;   // H channel base
    const int col = lane & 15, hi = lane >> 4, m = lane & 15;
    const bf16* arow0 = actw + (size_t)rows[m]      * I_DIM;
    const bf16* arow1 = actw + (size_t)rows[16 + m] * I_DIM;
    const bf16* wrow  = w2 + (size_t)e * H_DIM * I_DIM + (size_t)(n0 + col) * I_DIM;

    v8f acc0 = {}, acc1 = {};
    for (int k0 = 0; k0 < I_DIM; k0 += 32) {
        union { v16bf v; v8bf h[2]; } A0, A1, Bm;
        A0.h[0] = *(const v8bf*)&arow0[k0 + hi * 8];
        A0.h[1] = *(const v8bf*)&arow0[k0 + 16 + hi * 8];
        A1.h[0] = *(const v8bf*)&arow1[k0 + hi * 8];
        A1.h[1] = *(const v8bf*)&arow1[k0 + 16 + hi * 8];
        Bm.h[0] = *(const v8bf*)&wrow[k0 + hi * 8];
        Bm.h[1] = *(const v8bf*)&wrow[k0 + 16 + hi * 8];
        acc0 = __builtin_amdgcn_wmma_f32_16x16x32_bf16(
                   false, A0.v, false, Bm.v, (short)0, acc0, false, false);
        acc1 = __builtin_amdgcn_wmma_f32_16x16x32_bf16(
                   false, A1.v, false, Bm.v, (short)0, acc1, false, false);
    }
#pragma unroll
    for (int tile = 0; tile < 2; ++tile) {
        v8f acc = tile ? acc1 : acc0;
#pragma unroll
        for (int v = 0; v < 8; ++v) {
            int mrow = tile * 16 + v + hi * 8;
            if (base + mrow < cnt) {
                int token = rows[mrow] >> 1;
                __hip_atomic_fetch_add(&out_acc[(size_t)token * H_DIM + n0 + col],
                                       acc[v], __ATOMIC_RELAXED,
                                       __HIP_MEMORY_SCOPE_AGENT);
            }
        }
    }
}

// ---------------------------------------------------------------------------
// Kernel 4: out = bf16(x + acc + w0*b2[e0] + w1*b2[e1])
// ---------------------------------------------------------------------------
__global__ void __launch_bounds__(256) k_finalize(
    const bf16* __restrict__ x, const float* __restrict__ out_acc,
    const bf16* __restrict__ b2, const int* __restrict__ idx,
    const float* __restrict__ wgt, bf16* __restrict__ out)
{
    const int b = blockIdx.x, tid = threadIdx.x;
    const int i0 = idx[b * 2], i1 = idx[b * 2 + 1];
    const float w0 = wgt[b * 2], w1 = wgt[b * 2 + 1];
#pragma unroll
    for (int j = 0; j < 4; ++j) {
        int c = tid * 4 + j;
        float v = out_acc[(size_t)b * H_DIM + c]
                + w0 * (float)b2[(size_t)i0 * H_DIM + c]
                + w1 * (float)b2[(size_t)i1 * H_DIM + c]
                + (float)x[(size_t)b * H_DIM + c];
        out[(size_t)b * H_DIM + c] = (bf16)v;
    }
}

// ---------------------------------------------------------------------------
extern "C" void kernel_launch(void* const* d_in, const int* in_sizes, int n_in,
                              void* d_out, int out_size, void* d_ws, size_t ws_size,
                              hipStream_t stream)
{
    const bf16*  x      = (const bf16*) d_in[0];
    const float* nscale = (const float*)d_in[1];
    const bf16*  gw     = (const bf16*) d_in[2];
    const bf16*  gb     = (const bf16*) d_in[3];
    const bf16*  w1     = (const bf16*) d_in[4];
    const bf16*  b1     = (const bf16*) d_in[5];
    const bf16*  w2     = (const bf16*) d_in[6];
    const bf16*  b2     = (const bf16*) d_in[7];
    bf16* out = (bf16*)d_out;

    char* ws = (char*)d_ws;
    size_t off = 0;
    bf16*  t      = (bf16*)(ws + off); off += (size_t)B_NUM * H_DIM * 2;       // 2 MB
    int*   idx    = (int*)  (ws + off); off += (size_t)B_NUM * 2 * 4;          // 8 KB
    float* wgt    = (float*)(ws + off); off += (size_t)B_NUM * 2 * 4;          // 8 KB
    int*   ecount = (int*)  (ws + off); off += 256;
    int*   elist  = (int*)  (ws + off); off += (size_t)E_NUM * B_NUM * 4;      // 64 KB
    bf16*  actw   = (bf16*) (ws + off); off += (size_t)B_NUM * 2 * I_DIM * 2;  // 8 MB
    float* oacc   = (float*)(ws + off); off += (size_t)B_NUM * H_DIM * 4;      // 4 MB

    (void)hipMemsetAsync(ecount, 0, 256, stream);
    (void)hipMemsetAsync(oacc, 0, (size_t)B_NUM * H_DIM * 4, stream);

    k_norm_gate<<<B_NUM, 256, 0, stream>>>(x, nscale, gw, gb, t, idx, wgt, ecount, elist);
    k_mlp1_swiglu<<<dim3(TWO_I / 128, B_NUM / 32, E_NUM), 256, 0, stream>>>(
        t, w1, b1, wgt, ecount, elist, actw);
    k_mlp2<<<dim3(H_DIM / 128, B_NUM / 32, E_NUM), 256, 0, stream>>>(
        actw, w2, ecount, elist, oacc);
    k_finalize<<<B_NUM, 256, 0, stream>>>(x, oacc, b2, idx, wgt, out);
}
